// QuantizedLinear_30142080483927
// MI455X (gfx1250) — matrix operand order repaired and checked by reading, not compiled, not run on hardware
//
#include <hip/hip_runtime.h>
#include <hip/hip_bf16.h>

// Problem constants (from the reference)
#define M_DIM 8192      // 4 * 2048 tokens
#define K_DIM 4096      // IN_FEATURES
#define N_DIM 16384     // OUT_FEATURES

typedef __attribute__((ext_vector_type(16), aligned(16))) _Float16 v16h;
typedef __attribute__((ext_vector_type(8)))  _Float16 v8h;
typedef __attribute__((ext_vector_type(8)))  float    v8f;
typedef __attribute__((ext_vector_type(4)))  float    v4f;
typedef __attribute__((ext_vector_type(4)))  int      v4i;

// ---------------------------------------------------------------------------
// Conversion pass 1: x f32 -> f16  (33.5M elements, 8 per thread)
// ---------------------------------------------------------------------------
__global__ void __launch_bounds__(256)
cvt_x_kernel(const float* __restrict__ x, _Float16* __restrict__ xh) {
    size_t i = ((size_t)blockIdx.x * 256 + threadIdx.x) * 8;
    v4f a = __builtin_nontemporal_load((const v4f*)(x + i));
    v4f b = __builtin_nontemporal_load((const v4f*)(x + i + 4));
    v8h h;
    h[0] = (_Float16)a[0]; h[1] = (_Float16)a[1];
    h[2] = (_Float16)a[2]; h[3] = (_Float16)a[3];
    h[4] = (_Float16)b[0]; h[5] = (_Float16)b[1];
    h[6] = (_Float16)b[2]; h[7] = (_Float16)b[3];
    *(v8h*)(xh + i) = h;   // xh is re-read many times -> keep cacheable
}

// ---------------------------------------------------------------------------
// Conversion pass 2: q_weight i32 -> f16 (exact: |q| <= 127), 8 per thread
// Scale folded into the f32 epilogue, so dequantization stays exact.
// ---------------------------------------------------------------------------
__global__ void __launch_bounds__(256)
cvt_w_kernel(const int* __restrict__ q, _Float16* __restrict__ wh) {
    size_t i = ((size_t)blockIdx.x * 256 + threadIdx.x) * 8;
    v4i a = __builtin_nontemporal_load((const v4i*)(q + i));
    v4i b = __builtin_nontemporal_load((const v4i*)(q + i + 4));
    v8h h;
    h[0] = (_Float16)a[0]; h[1] = (_Float16)a[1];
    h[2] = (_Float16)a[2]; h[3] = (_Float16)a[3];
    h[4] = (_Float16)b[0]; h[5] = (_Float16)b[1];
    h[6] = (_Float16)b[2]; h[7] = (_Float16)b[3];
    *(v8h*)(wh + i) = h;
}

// ---------------------------------------------------------------------------
// WMMA GEMM: out[m,n] = scale * sum_k xh[m,k]*wh[n,k] + bias[n]
//   Block tile: 128(M) x 256(N), BK=32 per stage, double-buffered LDS (60KB).
//   8 waves (wave32) as 2(M) x 4(N); wave tile = 64x64 = 4x4 WMMA tiles
//   -> 16 v_wmma per 16 ds_load_b128 per stage.
//   Staging uses GLOBAL_LOAD_ASYNC_TO_LDS_B128 (ASYNCcnt-tracked DMA):
//   no staging VGPRs, no ds_store round trip -> fits 256-VGPR window.
// ---------------------------------------------------------------------------
#define BM 128
#define BN 256
#define BK 32
#define LDSS 40   // LDS row stride in halves (32 + 8 pad -> conflict-free)

__global__ void __launch_bounds__(256)
gemm_f16_wmma_kernel(const _Float16* __restrict__ A,   // [M][K] row-major
                     const _Float16* __restrict__ B,   // [N][K] row-major
                     const float* __restrict__ scale_p,
                     const float* __restrict__ bias,
                     float* __restrict__ C) {          // [M][N]
    __shared__ _Float16 sA[2][BM * LDSS];
    __shared__ _Float16 sB[2][BN * LDSS];

    const int tid   = threadIdx.x;
    const int lane  = tid & 31;
    const int wid   = tid >> 5;
    const int waveM = wid & 1;        // 0..1 -> 64-row slab
    const int waveN = wid >> 1;       // 0..3 -> 64-col slab
    const int m0    = blockIdx.x * BM;
    const int n0    = blockIdx.y * BN;

    // Staging: chunks of 8 halves (16B). A tile = 512 chunks (2/thread),
    // B tile = 1024 chunks (4/thread). 4 chunks per 32-half row.
    const int srow = tid >> 2;          // 0..63
    const int scol = (tid & 3) * 8;     // 0,8,16,24

    // Uniform (SGPR) bases, bumped by BK per stage; per-lane 32-bit BYTE
    // offsets are loop-invariant.
    const _Float16* pA = A + (size_t)m0 * K_DIM;
    const _Float16* pB = B + (size_t)n0 * K_DIM;
    const unsigned aByte0 = (unsigned)(((srow     ) * K_DIM + scol) * 2);
    const unsigned aByte1 = (unsigned)(((srow + 64) * K_DIM + scol) * 2);
    unsigned bByte[4];
    #pragma unroll
    for (int c = 0; c < 4; ++c)
        bByte[c] = (unsigned)(((c * 64 + srow) * K_DIM + scol) * 2);

    // 32-bit LDS addresses: LDS aperture occupies only the high 32 bits of a
    // generic pointer, so truncation yields the LDS byte address.
    const unsigned aLds = (unsigned)(unsigned long long)(&sA[0][srow * LDSS + scol]);
    const unsigned bLds = (unsigned)(unsigned long long)(&sB[0][srow * LDSS + scol]);

    // Async DMA staging: 6 x GLOBAL_LOAD_ASYNC_TO_LDS_B128 per wave per stage
    // (8 waves x 32 lanes x 6 x 16B = 24KB = A tile (8KB) + B tile (16KB)).
    auto issue_stage = [&](int bsel) {
        const unsigned long long gA = (unsigned long long)pA;
        const unsigned long long gB = (unsigned long long)pB;
        const unsigned la = aLds + (unsigned)bsel * (unsigned)(BM * LDSS * 2);
        const unsigned lb = bLds + (unsigned)bsel * (unsigned)(BN * LDSS * 2);
        asm volatile("global_load_async_to_lds_b128 %0, %1, %2"
                     :: "v"(la), "v"(aByte0), "s"(gA) : "memory");
        asm volatile("global_load_async_to_lds_b128 %0, %1, %2"
                     :: "v"(la + (unsigned)(64 * LDSS * 2)), "v"(aByte1), "s"(gA) : "memory");
        #pragma unroll
        for (int c = 0; c < 4; ++c)
            asm volatile("global_load_async_to_lds_b128 %0, %1, %2"
                         :: "v"(lb + (unsigned)(c * 64 * LDSS * 2)), "v"(bByte[c]), "s"(gB)
                         : "memory");
    };

    // Fragment addressing per CDNA5 WMMA VGPR layouts (wave32):
    //  A 16x32 f16: lane<16 -> M=lane,    halves {K 0..7,  K 16..23}
    //               lane>=16-> M=lane-16, halves {K 8..15, K 24..31}
//  B 32x16 f16: lane<16 -> N=lane,    halves {K 0..7,  K 16..23}
    //               lane>=16-> N=lane-16, halves {K 8..15, K 24..31}
    const int halfSel = lane >> 4;        // 0/1
    const int lrow    = lane & 15;
    const int akb     = halfSel * 8;      // A K base
    const int bkb     = halfSel * 8;      // B K base

    v8f acc[4][4] = {};                   // 4 M-tiles x 4 N-tiles, f32 accum

    auto compute_stage = [&](int b) {
        v16h bf[4];
        #pragma unroll
        for (int nt = 0; nt < 4; ++nt) {
            const _Float16* p = &sB[b][(waveN * 64 + nt * 16 + lrow) * LDSS + bkb];
            union { v16h v; v8h h[2]; } ub;
            ub.h[0] = *(const v8h*)p;         // K = bkb .. bkb+7
            ub.h[1] = *(const v8h*)(p + 16);  // K = bkb+16 .. bkb+23
            bf[nt] = ub.v;
        }
        #pragma unroll
        for (int mt = 0; mt < 4; ++mt) {
            const _Float16* p = &sA[b][(waveM * 64 + mt * 16 + lrow) * LDSS + akb];
            union { v16h v; v8h h[2]; } u;
            u.h[0] = *(const v8h*)p;          // K = akb .. akb+7
            u.h[1] = *(const v8h*)(p + 16);   // K = akb+16 .. akb+23
            const v16h af = u.v;
            #pragma unroll
            for (int nt = 0; nt < 4; ++nt)
                acc[mt][nt] = __builtin_amdgcn_wmma_f32_16x16x32_f16(
                    false, af, false, bf[nt],
                    (short)0, acc[mt][nt], false, false);
        }
    };

    // Pipeline: async-DMA stage s+1 into buf^1 while computing stage s from
    // buf; s_wait_asynccnt 0 (own transfers landed) + barrier (publish).
    const int S = K_DIM / BK;             // 128 stages
    issue_stage(0);
    asm volatile("s_wait_asynccnt 0x0" ::: "memory");
    __syncthreads();
    pA += BK; pB += BK;
    int buf = 0;
    for (int s = 0; s < S - 1; ++s) {
        issue_stage(buf ^ 1);             // DMA next stage
        compute_stage(buf);               // 16 WMMAs over current stage
        asm volatile("s_wait_asynccnt 0x0" ::: "memory");
        __syncthreads();
        pA += BK; pB += BK;               // uniform SALU bump only
        buf ^= 1;
    }
    compute_stage(buf);

    // Epilogue: C layout -> row = tile + v + 8*halfSel, col = tile + lrow.
    // Output is write-once (512MB): non-temporal stores keep L2 for xh.
    const float sc = *scale_p;
    #pragma unroll
    for (int nt = 0; nt < 4; ++nt) {
        const int col = n0 + waveN * 64 + nt * 16 + lrow;
        const float bv = bias[col];
        #pragma unroll
        for (int mt = 0; mt < 4; ++mt) {
            const int rbase = m0 + waveM * 64 + mt * 16 + halfSel * 8;
            #pragma unroll
            for (int v = 0; v < 8; ++v) {
                __builtin_nontemporal_store(
                    sc * acc[mt][nt][v] + bv,
                    C + (size_t)(rbase + v) * N_DIM + col);
            }
        }
    }
}

// ---------------------------------------------------------------------------
// Launch: convert -> GEMM. Workspace layout:
//   [0, 64MB)        xh : f16 activations (resident in 192MB L2 during GEMM)
//   [64MB, 192MB)    wh : f16 weights (streamed from HBM ~once thanks to
//                         M-fastest workgroup ordering within each N-band)
// ---------------------------------------------------------------------------
extern "C" void kernel_launch(void* const* d_in, const int* in_sizes, int n_in,
                              void* d_out, int out_size, void* d_ws, size_t ws_size,
                              hipStream_t stream) {
    const float* x     = (const float*)d_in[0];
    const int*   qw    = (const int*)  d_in[1];
    const float* scale = (const float*)d_in[2];
    const float* bias  = (const float*)d_in[3];
    float*       out   = (float*)d_out;

    _Float16* xh = (_Float16*)d_ws;
    _Float16* wh = (_Float16*)((char*)d_ws + (size_t)M_DIM * K_DIM * sizeof(_Float16));

    // x: 33,554,432 elems -> 8/thread -> 16384 blocks of 256
    cvt_x_kernel<<<dim3((M_DIM * (size_t)K_DIM) / (8 * 256)), dim3(256), 0, stream>>>(x, xh);
    // w: 67,108,864 elems -> 32768 blocks of 256
    cvt_w_kernel<<<dim3((N_DIM * (size_t)K_DIM) / (8 * 256)), dim3(256), 0, stream>>>(qw, wh);

    // grid.x = M tiles (fastest) so all M-blocks of an N-band run together:
    // weight band streams once, xh served from L2.
    dim3 grid(M_DIM / BM, N_DIM / BN);   // (64, 64)
    gemm_f16_wmma_kernel<<<grid, dim3(256), 0, stream>>>(xh, wh, scale, bias, out);
}
